// TranAD_TNT_AutoDIS_SelfAtt_LSTM_ASSA_TOP_M_63702954934612
// MI455X (gfx1250) — matrix operand order repaired and checked by reading, not compiled, not run on hardware
//
#include <hip/hip_runtime.h>
#include <hip/hip_bf16.h>
#include <math.h>

#define B_   256
#define F_   51
#define W_   9
#define WS_  8          // GRU steps (seq[:-1])
#define T_   459        // W_*F_
#define EMB_ 6
#define HID_ 306
#define NG_  918        // 3*HID
#define KP_  320        // K padded (306 -> 320, 10 wmma k-chunks)
#define NP_  960        // N padded (918 -> 960, 60 n-tiles = 15 groups of 4)
#define MGI_ 2048       // WS_*B_
#define TOPM 80

typedef __attribute__((ext_vector_type(16))) _Float16 v16h;
typedef __attribute__((ext_vector_type(8)))  _Float16 v8h;
typedef __attribute__((ext_vector_type(8)))  float    v8f;

// ---------------- utility ----------------
__global__ void zero_f16_kernel(_Float16* p, int n) {
    int i = blockIdx.x * blockDim.x + threadIdx.x;
    if (i < n) p[i] = (_Float16)0.0f;
}

// pack f32 weight (N x K row-major) into padded f16 (NP_ x KP_)
__global__ void pack_w16_kernel(const float* __restrict__ W, _Float16* __restrict__ W16,
                                int N, int K) {
    int i = blockIdx.x * blockDim.x + threadIdx.x;
    if (i >= NP_ * KP_) return;
    int n = i / KP_, k = i % KP_;
    float v = (n < N && k < K) ? W[n * K + k] : 0.0f;
    W16[i] = (_Float16)v;
}

// ---------------- 1) AutoDIS embedding ----------------
__global__ void embed_kernel(const float* __restrict__ src,
                             const float* __restrict__ w1_w, const float* __restrict__ w1_b,
                             const float* __restrict__ w2_w, const float* __restrict__ w2_b,
                             const float* __restrict__ meta, float* __restrict__ xbuf) {
    int idx = blockIdx.x * blockDim.x + threadIdx.x;
    if (idx >= W_ * B_ * F_) return;
    int w = idx / (B_ * F_);
    int r = idx % (B_ * F_);
    int b = r / F_, f = r % F_;
    float xv = src[(w * B_ + b) * F_ + f];

    float h1[EMB_], h2[EMB_];
#pragma unroll
    for (int i = 0; i < EMB_; ++i) {
        float t = xv * w1_w[f * EMB_ + i] + w1_b[f * EMB_ + i];
        h1[i] = (t > 0.0f) ? t : 0.01f * t;              // leaky_relu
    }
#pragma unroll
    for (int j = 0; j < EMB_; ++j) {
        float acc = w2_b[f * EMB_ + j];
#pragma unroll
        for (int i = 0; i < EMB_; ++i)
            acc += h1[i] * w2_w[(f * EMB_ + j) * EMB_ + i];
        h2[j] = acc;
    }
    float lg[EMB_], mx = -3.0e38f;
#pragma unroll
    for (int j = 0; j < EMB_; ++j) { lg[j] = h2[j] + 0.5f * h1[j]; mx = fmaxf(mx, lg[j]); }
    float pr[EMB_], sum = 0.0f;
#pragma unroll
    for (int j = 0; j < EMB_; ++j) { pr[j] = __expf((lg[j] - mx) * 1.0e5f); sum += pr[j]; }
    float inv = 1.0f / sum;
    const float sq = 2.44948974968f;                     // sqrt(6)
    float* xp = xbuf + ((size_t)b * T_ + (w * F_ + f)) * EMB_;
#pragma unroll
    for (int e = 0; e < EMB_; ++e) {
        float acc = 0.0f;
#pragma unroll
        for (int j = 0; j < EMB_; ++j)
            acc += pr[j] * inv * meta[(f * EMB_ + j) * EMB_ + e];
        xp[e] = acc * sq;
    }
}

// ---------------- 2) QKV projection ----------------
__global__ void qkv_kernel(const float* __restrict__ xbuf,
                           const float* __restrict__ Wq, const float* __restrict__ bq,
                           const float* __restrict__ Wk, const float* __restrict__ bk,
                           const float* __restrict__ Wv, const float* __restrict__ bv,
                           float* __restrict__ Q, float* __restrict__ K, float* __restrict__ V) {
    int idx = blockIdx.x * blockDim.x + threadIdx.x;
    if (idx >= B_ * T_) return;
    int b = idx / T_, t = idx % T_;
    float x[EMB_];
#pragma unroll
    for (int e = 0; e < EMB_; ++e) x[e] = xbuf[((size_t)b * T_ + t) * EMB_ + e];
#pragma unroll
    for (int o = 0; o < EMB_; ++o) {
        float aq = bq[o], ak = bk[o], av = bv[o];
#pragma unroll
        for (int e = 0; e < EMB_; ++e) {
            aq += x[e] * Wq[o * EMB_ + e];
            ak += x[e] * Wk[o * EMB_ + e];
            av += x[e] * Wv[o * EMB_ + e];
        }
        int h = o >> 1, d = o & 1;                       // (B,H,T,D)
        size_t p = (((size_t)b * 3 + h) * T_ + t) * 2 + d;
        Q[p] = aq; K[p] = ak; V[p] = av;
    }
}

// ---------------- 3) sparse-top-M attention, one wave per (b,h,row) --------
__global__ __launch_bounds__(32) void attn_kernel(const float* __restrict__ Q,
                                                  const float* __restrict__ K,
                                                  const float* __restrict__ V,
                                                  float* __restrict__ O) {
    int gid  = blockIdx.x;
    int qrow = gid % T_;
    int bh   = gid / T_;                 // b*3 + h
    int lane = threadIdx.x;
    const float rsq = 0.70710678118f;    // 1/sqrt(DHEAD)

    const float* qp = Q + ((size_t)bh * T_ + qrow) * 2;
    float q0 = qp[0], q1 = qp[1];
    const float* kb = K + (size_t)bh * T_ * 2;

    float    sc[15];
    unsigned su[15];
#pragma unroll
    for (int j = 0; j < 15; ++j) {
        int kk = lane + 32 * j;
        float s = -3.0e38f;
        if (kk < T_) s = (q0 * kb[kk * 2] + q1 * kb[kk * 2 + 1]) * rsq;
        sc[j] = s;
        unsigned bits = __float_as_uint(s);
        su[j] = (kk < T_) ? ((bits & 0x80000000u) ? ~bits : (bits | 0x80000000u)) : 0u;
    }
    // exact radix-select of the TOPM-th largest value (MSB-first)
    unsigned prefix = 0; int kneed = TOPM;
    for (int bit = 31; bit >= 0; --bit) {
        unsigned cand = prefix | (1u << bit);
        unsigned mask = ~((1u << bit) - 1u);
        int c = 0;
#pragma unroll
        for (int j = 0; j < 15; ++j) c += ((su[j] & mask) == cand) ? 1 : 0;
        for (int off = 16; off; off >>= 1) c += __shfl_xor(c, off, 32);
        if (c >= kneed) prefix = cand; else kneed -= c;
    }
    unsigned tb = (prefix & 0x80000000u) ? (prefix ^ 0x80000000u) : ~prefix;
    float kth = __uint_as_float(tb);

    float mx = -3.0e38f;
#pragma unroll
    for (int j = 0; j < 15; ++j) mx = fmaxf(mx, sc[j]);
    for (int off = 16; off; off >>= 1) mx = fmaxf(mx, __shfl_xor(mx, off, 32));

    const float* vb = V + (size_t)bh * T_ * 2;
    float sum = 0.0f, o0 = 0.0f, o1 = 0.0f;
#pragma unroll
    for (int j = 0; j < 15; ++j) {
        int kk = lane + 32 * j;
        if (kk < T_ && sc[j] >= kth) {
            float e = __expf(sc[j] - mx);
            sum += e; o0 += e * vb[kk * 2]; o1 += e * vb[kk * 2 + 1];
        }
    }
    for (int off = 16; off; off >>= 1) {
        sum += __shfl_xor(sum, off, 32);
        o0  += __shfl_xor(o0,  off, 32);
        o1  += __shfl_xor(o1,  off, 32);
    }
    if (lane == 0) {
        int h = bh % 3, b = bh / 3;
        float inv = 1.0f / sum;
        float* op = O + ((size_t)b * T_ + qrow) * EMB_ + 2 * h;   // (B,T,EMB)
        op[0] = o0 * inv; op[1] = o1 * inv;
    }
}

// ---------------- 4) Wo + LN1 + FFN + LN2, emit fp16 GRU input -------------
__global__ void post_kernel(const float* __restrict__ xbuf, const float* __restrict__ O,
                            const float* __restrict__ Wo, const float* __restrict__ bo,
                            const float* __restrict__ l1s, const float* __restrict__ l1b,
                            const float* __restrict__ fw1, const float* __restrict__ fb1,
                            const float* __restrict__ fw2, const float* __restrict__ fb2,
                            const float* __restrict__ l2s, const float* __restrict__ l2b,
                            _Float16* __restrict__ A16) {
    int idx = blockIdx.x * blockDim.x + threadIdx.x;
    if (idx >= B_ * WS_ * F_) return;
    int b = idx / (WS_ * F_);
    int r = idx % (WS_ * F_);
    int w = r / F_, f = r % F_;
    int t = w * F_ + f;

    float x[EMB_], o6[EMB_], y[EMB_];
#pragma unroll
    for (int e = 0; e < EMB_; ++e) {
        x[e]  = xbuf[((size_t)b * T_ + t) * EMB_ + e];
        o6[e] = O[((size_t)b * T_ + t) * EMB_ + e];
    }
    // x + o @ Wo^T + bo
#pragma unroll
    for (int o = 0; o < EMB_; ++o) {
        float a = bo[o];
#pragma unroll
        for (int e = 0; e < EMB_; ++e) a += o6[e] * Wo[o * EMB_ + e];
        y[o] = x[o] + a;
    }
    // LN1
    float mu = 0.0f;
#pragma unroll
    for (int e = 0; e < EMB_; ++e) mu += y[e];
    mu *= (1.0f / EMB_);
    float var = 0.0f;
#pragma unroll
    for (int e = 0; e < EMB_; ++e) { float d = y[e] - mu; var += d * d; }
    var *= (1.0f / EMB_);
    float rs = rsqrtf(var + 1e-5f);
    float x1[EMB_];
#pragma unroll
    for (int e = 0; e < EMB_; ++e) x1[e] = (y[e] - mu) * rs * l1s[e] + l1b[e];
    // FFN
    float f1[12];
#pragma unroll
    for (int k = 0; k < 12; ++k) {
        float a = fb1[k];
#pragma unroll
        for (int e = 0; e < EMB_; ++e) a += x1[e] * fw1[k * EMB_ + e];
        f1[k] = fmaxf(a, 0.0f);
    }
#pragma unroll
    for (int e = 0; e < EMB_; ++e) {
        float a = fb2[e];
#pragma unroll
        for (int k = 0; k < 12; ++k) a += f1[k] * fw2[e * 12 + k];
        y[e] = x1[e] + a;
    }
    // LN2
    mu = 0.0f;
#pragma unroll
    for (int e = 0; e < EMB_; ++e) mu += y[e];
    mu *= (1.0f / EMB_);
    var = 0.0f;
#pragma unroll
    for (int e = 0; e < EMB_; ++e) { float d = y[e] - mu; var += d * d; }
    var *= (1.0f / EMB_);
    rs = rsqrtf(var + 1e-5f);
    // seq[w][b][f*6+e] -> fp16 GRU-input row (w*B_+b), col f*6+e
    _Float16* ap = A16 + ((size_t)(w * B_ + b)) * KP_ + f * EMB_;
#pragma unroll
    for (int e = 0; e < EMB_; ++e)
        ap[e] = (_Float16)((y[e] - mu) * rs * l2s[e] + l2b[e]);
}

// ------- 5) WMMA f16 GEMM, 16x64 per wave: C[M,NP_] = A[M,KP_] x W[N,KP_]^T
__global__ __launch_bounds__(256) void gemm_f16_wmma(const _Float16* __restrict__ A,
                                                     const _Float16* __restrict__ Wt,
                                                     float* __restrict__ C,
                                                     int Mtiles, int Ngroups) {
    int wave = blockIdx.x * (blockDim.x >> 5) + (threadIdx.x >> 5);
    if (wave >= Mtiles * Ngroups) return;           // whole-wave uniform exit
    int lane = threadIdx.x & 31;
    int mt = wave / Ngroups, ng = wave % Ngroups;
    int half = lane >> 4, l16 = lane & 15;

    const _Float16* arow = A  + (size_t)(mt * 16 + l16) * KP_;
    const _Float16* brow = Wt + (size_t)(ng * 64 + l16) * KP_;   // n-tile 0 of group

    v8f acc[4] = {};
#pragma unroll
    for (int kc = 0; kc < KP_; kc += 32) {
        // A frag (16x32 f16): lanes 0-15 K {0..7,16..23}; lanes 16-31 K {8..15,24..31}
        int ao = kc + half * 8;
        v8h alo = *(const v8h*)(arow + ao);
        v8h ahi = *(const v8h*)(arow + ao + 16);
        v16h a;
#pragma unroll
        for (int i = 0; i < 8; ++i) { a[i] = alo[i]; a[i + 8] = ahi[i]; }
        // 4 B tiles reuse the same A fragment (16x64 register blocking)
        int bo = kc + half * 8;
#pragma unroll
        for (int u = 0; u < 4; ++u) {
            const _Float16* bp = brow + (size_t)u * 16 * KP_ + bo;  // immediate offsets
            v8h blo = *(const v8h*)bp;
            v8h bhi = *(const v8h*)(bp + 16);
            v16h bm;
#pragma unroll
            for (int i = 0; i < 8; ++i) { bm[i] = blo[i]; bm[i + 8] = bhi[i]; }
            acc[u] = __builtin_amdgcn_wmma_f32_16x16x32_f16(false, a, false, bm,
                                                            (short)0, acc[u], false, false);
        }
    }
    // C layout: VGPR d -> row M = d + half*8; col N = l16
#pragma unroll
    for (int u = 0; u < 4; ++u) {
        float* cp = C + (size_t)(mt * 16 + half * 8) * NP_ + ng * 64 + u * 16 + l16;
#pragma unroll
        for (int d = 0; d < 8; ++d) cp[(size_t)d * NP_] = acc[u][d];
    }
}

// ---------------- 6) GRU state init / gate update / head ------------------
__global__ void hinit_kernel(const float* __restrict__ h0, float* __restrict__ h32,
                             _Float16* __restrict__ h16) {
    int idx = blockIdx.x * blockDim.x + threadIdx.x;
    if (idx >= B_ * KP_) return;
    int b = idx / KP_, j = idx % KP_;
    if (j < HID_) {
        float v = h0[(size_t)b * HID_ + j];
        h32[(size_t)b * HID_ + j] = v;
        h16[idx] = (_Float16)v;
    } else {
        h16[idx] = (_Float16)0.0f;
    }
}

__global__ void gru_gate_kernel(const float* __restrict__ Cgi, const float* __restrict__ Cgh,
                                const float* __restrict__ bih, const float* __restrict__ bhh,
                                float* __restrict__ h32, _Float16* __restrict__ h16, int w) {
    int idx = blockIdx.x * blockDim.x + threadIdx.x;
    if (idx >= B_ * HID_) return;
    int b = idx / HID_, j = idx % HID_;
    const float* gi = Cgi + (size_t)(w * B_ + b) * NP_;
    const float* gh = Cgh + (size_t)b * NP_;
    float ir = gi[j]            + bih[j];
    float iz = gi[HID_ + j]     + bih[HID_ + j];
    float in = gi[2 * HID_ + j] + bih[2 * HID_ + j];
    float hr = gh[j]            + bhh[j];
    float hz = gh[HID_ + j]     + bhh[HID_ + j];
    float hn = gh[2 * HID_ + j] + bhh[2 * HID_ + j];
    float rg = 1.0f / (1.0f + __expf(-(ir + hr)));
    float zg = 1.0f / (1.0f + __expf(-(iz + hz)));
    float ng = tanhf(in + rg * hn);
    float hp = h32[(size_t)b * HID_ + j];
    float hnew = (1.0f - zg) * ng + zg * hp;
    h32[(size_t)b * HID_ + j] = hnew;
    h16[(size_t)b * KP_ + j]  = (_Float16)hnew;
}

__global__ void fcn_kernel(const float* __restrict__ h32, const float* __restrict__ fw,
                           const float* __restrict__ fb, float* __restrict__ out) {
    int idx = blockIdx.x * blockDim.x + threadIdx.x;
    if (idx >= B_ * F_) return;
    int b = idx / F_, f = idx % F_;
    float a = fb[f];
    const float* hr = h32 + (size_t)b * HID_;
    const float* wr = fw + (size_t)f * HID_;
    for (int j = 0; j < HID_; ++j) a += hr[j] * wr[j];
    out[idx] = 1.0f / (1.0f + __expf(-a));
}

// ---------------- launch ----------------
extern "C" void kernel_launch(void* const* d_in, const int* in_sizes, int n_in,
                              void* d_out, int out_size, void* d_ws, size_t ws_size,
                              hipStream_t stream) {
    const float* src   = (const float*)d_in[0];
    const float* w1_w  = (const float*)d_in[2];
    const float* w1_b  = (const float*)d_in[3];
    const float* w2_w  = (const float*)d_in[4];
    const float* w2_b  = (const float*)d_in[5];
    const float* meta  = (const float*)d_in[6];
    const float* Wq    = (const float*)d_in[7];
    const float* bq    = (const float*)d_in[8];
    const float* Wk    = (const float*)d_in[9];
    const float* bk    = (const float*)d_in[10];
    const float* Wv    = (const float*)d_in[11];
    const float* bv    = (const float*)d_in[12];
    const float* Wo    = (const float*)d_in[13];
    const float* bo    = (const float*)d_in[14];
    const float* ln1_s = (const float*)d_in[15];
    const float* ln1_b = (const float*)d_in[16];
    const float* fw1   = (const float*)d_in[17];
    const float* fb1   = (const float*)d_in[18];
    const float* fw2   = (const float*)d_in[19];
    const float* fb2   = (const float*)d_in[20];
    const float* ln2_s = (const float*)d_in[21];
    const float* ln2_b = (const float*)d_in[22];
    const float* gwih  = (const float*)d_in[23];
    const float* gwhh  = (const float*)d_in[24];
    const float* gbih  = (const float*)d_in[25];
    const float* gbhh  = (const float*)d_in[26];
    const float* h0    = (const float*)d_in[27];
    const float* fcw   = (const float*)d_in[28];
    const float* fcb   = (const float*)d_in[29];

    char* ws = (char*)d_ws;
    size_t off = 0;
    auto alloc = [&](size_t bytes) -> char* {
        char* p = ws + off;
        off += (bytes + 255) & ~(size_t)255;
        return p;
    };
    float*     xbuf = (float*)    alloc((size_t)B_ * T_ * EMB_ * 4);
    float*     Q    = (float*)    alloc((size_t)B_ * 3 * T_ * 2 * 4);
    float*     Kb   = (float*)    alloc((size_t)B_ * 3 * T_ * 2 * 4);
    float*     Vb   = (float*)    alloc((size_t)B_ * 3 * T_ * 2 * 4);
    float*     O    = (float*)    alloc((size_t)B_ * T_ * EMB_ * 4);
    _Float16*  A16  = (_Float16*) alloc((size_t)MGI_ * KP_ * 2);
    _Float16*  W16i = (_Float16*) alloc((size_t)NP_ * KP_ * 2);
    _Float16*  W16h = (_Float16*) alloc((size_t)NP_ * KP_ * 2);
    float*     Cgi  = (float*)    alloc((size_t)MGI_ * NP_ * 4);
    float*     Cgh  = (float*)    alloc((size_t)B_ * NP_ * 4);
    float*     h32  = (float*)    alloc((size_t)B_ * HID_ * 4);
    _Float16*  h16  = (_Float16*) alloc((size_t)B_ * KP_ * 2);

    int n;
    n = MGI_ * KP_;
    zero_f16_kernel<<<(n + 255) / 256, 256, 0, stream>>>(A16, n);
    n = NP_ * KP_;
    pack_w16_kernel<<<(n + 255) / 256, 256, 0, stream>>>(gwih, W16i, NG_, HID_);
    pack_w16_kernel<<<(n + 255) / 256, 256, 0, stream>>>(gwhh, W16h, NG_, HID_);

    n = W_ * B_ * F_;
    embed_kernel<<<(n + 255) / 256, 256, 0, stream>>>(src, w1_w, w1_b, w2_w, w2_b, meta, xbuf);
    n = B_ * T_;
    qkv_kernel<<<(n + 255) / 256, 256, 0, stream>>>(xbuf, Wq, bq, Wk, bk, Wv, bv, Q, Kb, Vb);
    attn_kernel<<<B_ * 3 * T_, 32, 0, stream>>>(Q, Kb, Vb, O);
    n = B_ * WS_ * F_;
    post_kernel<<<(n + 255) / 256, 256, 0, stream>>>(xbuf, O, Wo, bo, ln1_s, ln1_b,
                                                     fw1, fb1, fw2, fb2, ln2_s, ln2_b, A16);

    // gi for all 8 steps at once: (2048 x 320) x (960 x 320)^T, 16x64 per wave
    {
        int waves = (MGI_ / 16) * (NP_ / 64);   // 128*15 = 1920 -> 240 blocks of 8 waves
        gemm_f16_wmma<<<waves / 8, 256, 0, stream>>>(A16, W16i, Cgi, MGI_ / 16, NP_ / 64);
    }
    n = B_ * KP_;
    hinit_kernel<<<(n + 255) / 256, 256, 0, stream>>>(h0, h32, h16);

    for (int w = 0; w < WS_; ++w) {
        int waves = (B_ / 16) * (NP_ / 64);     // 16*15 = 240 -> 30 blocks
        gemm_f16_wmma<<<waves / 8, 256, 0, stream>>>(h16, W16h, Cgh, B_ / 16, NP_ / 64);
        n = B_ * HID_;
        gru_gate_kernel<<<(n + 255) / 256, 256, 0, stream>>>(Cgi, Cgh, gbih, gbhh, h32, h16, w);
    }
    n = B_ * F_;
    fcn_kernel<<<(n + 255) / 256, 256, 0, stream>>>(h32, fcw, fcb, (float*)d_out);
}
